// EfficientMultiScaleDilatedAttention_9749575762711
// MI455X (gfx1250) — matrix operand order repaired and checked by reading, not compiled, not run on hardware
//
#include <hip/hip_runtime.h>
#include <hip/hip_bf16.h>
#include <stdint.h>

// ---------------------------------------------------------------------------
// EfficientMultiScaleDilatedAttention for MI455X (gfx1250), wave32 + WMMA bf16
// Pipeline: prep(weightsT->bf16) -> convert+pool(x) -> kv-proj -> E^T compress
//           -> Q-proj -> attention(ctx=80) -> out-proj + LayerNorm
// GEMM blocks are 32 rows x 768 cols (2 row-groups x 4 col-groups of waves)
// to halve L2 traffic on the transposed weights.
// ---------------------------------------------------------------------------

typedef unsigned short u16;
typedef __attribute__((ext_vector_type(16))) __bf16    v16bf;
typedef __attribute__((ext_vector_type(8)))  float     v8f;
typedef __attribute__((ext_vector_type(4)))  unsigned int u32x4;
typedef __attribute__((ext_vector_type(8)))  int       i32x8;
typedef __attribute__((ext_vector_type(4)))  int       i32x4;

#define B_      16
#define N_      4096
#define C_      768
#define H_      12
#define D_      64
#define LSEQ_   128
#define LK_     64
#define BANK_   16
#define KK_     80      /* LK_ + BANK_ */
#define KKP_    96      /* padded to multiple of 32 for WMMA K loop */

__device__ __forceinline__ u16 f2bf(float f) {
  union { float f; uint32_t u; } c; c.f = f;
  uint32_t r = c.u + 0x7FFFu + ((c.u >> 16) & 1u);   // round-to-nearest-even
  return (u16)(r >> 16);
}

__device__ __forceinline__ v8f v8f_zero() {
  v8f z = {0.f, 0.f, 0.f, 0.f, 0.f, 0.f, 0.f, 0.f};
  return z;
}

// A fragment (16x32 bf16, M x K): lane l holds row (l&15); 16 contiguous K
// at kstep*32 + (l>>4)*16 (caller passes koff = kstep*32 + (l>>4)*8).
__device__ __forceinline__ v16bf load_frag_a(const u16* row, int koff) {
  union { v16bf v; uint4 q[2]; } u;
  koff += (koff & 8);
  u.q[0] = *(const uint4*)(row + koff);
  u.q[1] = *(const uint4*)(row + koff + 8);
  return u.v;
}
// B fragment (32x16 bf16, K x N) from N-major (transposed) storage:
// lane l holds column (l&15); 32 contiguous bytes at koff = kstep*32 + (l>>4)*16.
__device__ __forceinline__ v16bf load_frag_b(const u16* rowT, int koff) {
  union { v16bf v; uint4 q[2]; } u;
  u.q[0] = *(const uint4*)(rowT + koff);
  u.q[1] = *(const uint4*)(rowT + koff + 8);
  return u.v;
}

__device__ __forceinline__ v8f wmma_bf16(v16bf a, v16bf b, v8f c) {
  return __builtin_amdgcn_wmma_f32_16x16x32_bf16(false, a, false, b, (short)0, c,
                                                 false, false);
}

// ---------------------------------------------------------------------------
// Tensor Data Mover: 1-D tile DMA global -> LDS (data_size = 8 bytes)
// ---------------------------------------------------------------------------
#if __has_builtin(__builtin_amdgcn_tensor_load_to_lds)
#define HAVE_TDM 1
#else
#define HAVE_TDM 0
#endif

__device__ __forceinline__ void tdm_load_lds(void* lds_dst, const void* gsrc,
                                             uint32_t nbytes) {
#if HAVE_TDM
  uint32_t lds_off = (uint32_t)(uintptr_t)lds_dst;       // low 32 bits = LDS offset
  uint64_t ga      = (uint64_t)(uintptr_t)gsrc;
  uint32_t units   = nbytes >> 3;                        // 8-byte elements
  u32x4 g0;
  g0[0] = 1u;                                            // count=1 (valid user D#)
  g0[1] = lds_off;                                       // lds_addr [63:32]
  g0[2] = (uint32_t)ga;                                  // global_addr [95:64]
  g0[3] = (uint32_t)((ga >> 32) & 0x01FFFFFFu) | 0x80000000u; // addr[56:32], type=2
  i32x8 g1;
  g1[0] = (int)(3u << 16);                               // data_size = 8B
  g1[1] = (int)((units & 0xFFFFu) << 16);                // tensor_dim0[15:0]
  g1[2] = (int)(((units >> 16) & 0xFFFFu) | (1u << 16)); // dim0[31:16], tensor_dim1=1
  g1[3] = (int)((units & 0xFFFFu) << 16);                // tile_dim0 = units
  g1[4] = 1;                                             // tile_dim1=1, tile_dim2=0
  g1[5] = (int)units;                                    // tensor_dim0_stride
  g1[6] = 0;
  g1[7] = 0;
  i32x4 z4 = {0, 0, 0, 0};
#if defined(__clang_major__) && (__clang_major__ >= 23)
  i32x8 z8 = {0, 0, 0, 0, 0, 0, 0, 0};
  __builtin_amdgcn_tensor_load_to_lds(g0, g1, z4, z4, z8, 0);
#else
  __builtin_amdgcn_tensor_load_to_lds(g0, g1, z4, z4, 0);
#endif
#else
  (void)lds_dst; (void)gsrc; (void)nbytes;
#endif
}

__device__ __forceinline__ void tdm_wait() {
#if HAVE_TDM
#if __has_builtin(__builtin_amdgcn_s_wait_tensorcnt)
  __builtin_amdgcn_s_wait_tensorcnt(0);
#else
  asm volatile("s_wait_tensorcnt 0x0" ::: "memory");
#endif
#endif
}

// ---------------------------------------------------------------------------
// Kernel 1: transpose + bf16-convert all weight matrices
// ---------------------------------------------------------------------------
__global__ void prep_weights_kernel(const float* __restrict__ Wqkv,
                                    const float* __restrict__ Wproj,
                                    const float* __restrict__ Ek,
                                    const float* __restrict__ Ev,
                                    u16* __restrict__ WqT, u16* __restrict__ WkvT,
                                    u16* __restrict__ WpT, u16* __restrict__ EkT,
                                    u16* __restrict__ EvT) {
  const int S1 = C_ * C_, S2 = 2 * C_ * C_, S3 = C_ * C_, S4 = LK_ * LSEQ_;
  int i = blockIdx.x * 256 + threadIdx.x;
  if (i < S1) { int n = i / C_, k = i % C_; WqT[i]  = f2bf(Wqkv[(size_t)k * (3*C_) + n]); return; }
  i -= S1;
  if (i < S2) { int n = i / C_, k = i % C_; WkvT[i] = f2bf(Wqkv[(size_t)k * (3*C_) + C_ + n]); return; }
  i -= S2;
  if (i < S3) { int n = i / C_, k = i % C_; WpT[i]  = f2bf(Wproj[(size_t)k * C_ + n]); return; }
  i -= S3;
  if (i < S4) { int kk = i / LSEQ_, n = i % LSEQ_; EkT[i] = f2bf(Ek[(size_t)n * LK_ + kk]); return; }
  i -= S4;
  if (i < S4) { int kk = i / LSEQ_, n = i % LSEQ_; EvT[i] = f2bf(Ev[(size_t)n * LK_ + kk]); return; }
}

// ---------------------------------------------------------------------------
// Kernel 2: x -> bf16; pooled first-128 tokens (pairwise mean of x[:,0:256])
// ---------------------------------------------------------------------------
__global__ void convert_pool_kernel(const float* __restrict__ x,
                                    u16* __restrict__ xbf,
                                    u16* __restrict__ xpbf) {
  const size_t NX = (size_t)B_ * N_ * C_;
  size_t i = (size_t)blockIdx.x * 256 + threadIdx.x;
  if (i < NX) {
    xbf[i] = f2bf(x[i]);
  } else {
    size_t j = i - NX;
    if (j < (size_t)B_ * LSEQ_ * C_) {
      int b = (int)(j / (LSEQ_ * C_));
      int r = (int)(j % (LSEQ_ * C_));
      int m = r / C_, c = r % C_;
      const float* p = x + ((size_t)b * N_ + 2 * m) * C_ + c;
      xpbf[j] = f2bf(0.5f * (p[0] + p[C_]));
    }
  }
}

// ---------------------------------------------------------------------------
// Kernel 3: pooled KV projection  kv[2048 x 1536] = xp @ W_qkv[:,768:] + b
// ---------------------------------------------------------------------------
__global__ void __launch_bounds__(256)
kvproj_kernel(const u16* __restrict__ xpbf, const u16* __restrict__ WkvT,
              const float* __restrict__ b_qkv, u16* __restrict__ kvbf) {
  __shared__ __align__(16) u16 Atile[16 * C_];
  const int row0 = blockIdx.x * 16;
  const u16* src = xpbf + (size_t)row0 * C_;
#if HAVE_TDM
  if (threadIdx.x < 32) { tdm_load_lds(Atile, src, 16 * C_ * 2); tdm_wait(); }
#else
  for (int i = threadIdx.x; i < 16 * C_; i += 256) Atile[i] = src[i];
#endif
  __syncthreads();

  const int lane = threadIdx.x & 31, w = threadIdx.x >> 5;
  const int mr = lane & 15, kh = lane >> 4;
  const u16* arow = Atile + mr * C_;

  v8f acc[12];
#pragma unroll
  for (int j = 0; j < 12; ++j) acc[j] = v8f_zero();

  for (int ks = 0; ks < C_ / 32; ++ks) {
    v16bf a = load_frag_a(arow, ks * 32 + kh * 8);
#pragma unroll
    for (int j = 0; j < 12; ++j) {
      const int n = (w * 12 + j) * 16 + mr;
      v16bf bm = load_frag_b(WkvT + (size_t)n * C_, ks * 32 + kh * 16);
      acc[j] = wmma_bf16(a, bm, acc[j]);
    }
  }
#pragma unroll
  for (int j = 0; j < 12; ++j) {
    const int col = (w * 12 + j) * 16 + mr;
    const float bias = b_qkv[C_ + col];
#pragma unroll
    for (int i = 0; i < 8; ++i) {
      const int M = (lane < 16) ? i : (i + 8);
      kvbf[(size_t)(row0 + M) * (2 * C_) + col] = f2bf(acc[j][i] + bias);
    }
  }
}

// ---------------------------------------------------------------------------
// Kernel 4: per-(b,h) compression  k_comp = E_k^T(64x128) @ K(128x64),
// v_comp likewise; append 16 bank rows; emit k_full[80x64] bf16 and
// v_full^T[64x96] bf16 (cols 80..95 zero so the attention P-pad is inert).
// ---------------------------------------------------------------------------
__global__ void __launch_bounds__(256)
kv_compress_kernel(const u16* __restrict__ kvbf, const u16* __restrict__ EkT,
                   const u16* __restrict__ EvT, const float* __restrict__ gk,
                   const float* __restrict__ gv, u16* __restrict__ kfull,
                   u16* __restrict__ vT) {
  __shared__ __align__(16) u16 KT[D_ * LSEQ_];   // KT[d][n] = K[n][d]
  __shared__ __align__(16) u16 VTs[D_ * LSEQ_];
  const int bh = blockIdx.x;
  const int b = bh / H_, h = bh % H_;
  for (int i = threadIdx.x; i < D_ * LSEQ_; i += 256) {
    const int d = i >> 7, n = i & 127;
    const u16* base = kvbf + (size_t)(b * LSEQ_ + n) * (2 * C_);
    KT[i]  = base[h * D_ + d];
    VTs[i] = base[C_ + h * D_ + d];
  }
  __syncthreads();

  const int lane = threadIdx.x & 31, w = threadIdx.x >> 5;
  const int mr = lane & 15, kh = lane >> 4;
  const int side = w >> 2;                    // 0 = K, 1 = V
  const int mt = w & 3;                       // kk tile
  const u16* ET = side ? EvT : EkT;
  const u16* T  = side ? VTs : KT;
  const u16* arow = ET + (size_t)(mt * 16 + mr) * LSEQ_;

  v8f acc[4];
#pragma unroll
  for (int j = 0; j < 4; ++j) acc[j] = v8f_zero();
#pragma unroll
  for (int ks = 0; ks < LSEQ_ / 32; ++ks) {
    v16bf a = load_frag_a(arow, ks * 32 + kh * 8);
#pragma unroll
    for (int j = 0; j < 4; ++j) {
      v16bf bm = load_frag_b(T + (size_t)(j * 16 + mr) * LSEQ_, ks * 32 + kh * 16);
      acc[j] = wmma_bf16(a, bm, acc[j]);
    }
  }
#pragma unroll
  for (int j = 0; j < 4; ++j) {
#pragma unroll
    for (int i = 0; i < 8; ++i) {
      const int M = (lane < 16) ? i : (i + 8);
      const int kk = mt * 16 + M, d = j * 16 + mr;
      if (side == 0)
        kfull[((size_t)bh * KK_ + kk) * D_ + d] = f2bf(acc[j][i]);
      else
        vT[((size_t)bh * D_ + d) * KKP_ + kk] = f2bf(acc[j][i]);
    }
  }
  // bank rows (broadcast global_k/global_v) + zero padding of vT cols 80..95
  for (int i = threadIdx.x; i < BANK_ * D_; i += 256) {
    const int j = i >> 6, d = i & 63;
    kfull[((size_t)bh * KK_ + LK_ + j) * D_ + d] = f2bf(gk[(size_t)j * C_ + h * D_ + d]);
    vT[((size_t)bh * D_ + d) * KKP_ + LK_ + j]   = f2bf(gv[(size_t)j * C_ + h * D_ + d]);
  }
  for (int i = threadIdx.x; i < (KKP_ - KK_) * D_; i += 256) {
    const int j = i >> 6, d = i & 63;
    vT[((size_t)bh * D_ + d) * KKP_ + KK_ + j] = 0;
  }
}

// ---------------------------------------------------------------------------
// Kernel 5: Q projection (65536 x 768 x 768), 32-row blocks, fold scale,
// write bf16 Q in (b, h, n, d) layout for contiguous attention tiles.
// 8 waves = 2 row-groups x 4 col-groups; 12 accumulators per wave.
// ---------------------------------------------------------------------------
__global__ void __launch_bounds__(256)
qproj_kernel(const u16* __restrict__ xbf, const u16* __restrict__ WqT,
             const float* __restrict__ b_qkv, u16* __restrict__ qbf) {
  __shared__ __align__(16) u16 Atile[32 * C_];
  const int row0 = blockIdx.x * 32;
  const u16* src = xbf + (size_t)row0 * C_;
#if HAVE_TDM
  if (threadIdx.x < 32) { tdm_load_lds(Atile, src, 32 * C_ * 2); tdm_wait(); }
#else
  for (int i = threadIdx.x; i < 32 * C_; i += 256) Atile[i] = src[i];
#endif
  __syncthreads();

  const int lane = threadIdx.x & 31, w = threadIdx.x >> 5;
  const int mr = lane & 15, kh = lane >> 4;
  const int rg = w >> 2;                      // row group (0..1)
  const int cg = w & 3;                       // col group (0..3)
  const u16* arow = Atile + (size_t)(rg * 16 + mr) * C_;

  v8f acc[12];
#pragma unroll
  for (int j = 0; j < 12; ++j) acc[j] = v8f_zero();

  for (int ks = 0; ks < C_ / 32; ++ks) {
    v16bf a = load_frag_a(arow, ks * 32 + kh * 8);
#pragma unroll
    for (int j = 0; j < 12; ++j) {
      const int n = (cg * 12 + j) * 16 + mr;
      v16bf bm = load_frag_b(WqT + (size_t)n * C_, ks * 32 + kh * 16);
      acc[j] = wmma_bf16(a, bm, acc[j]);
    }
  }
#pragma unroll
  for (int j = 0; j < 12; ++j) {
    const int col = (cg * 12 + j) * 16 + mr;
    const float bias = b_qkv[col];
    const int h = col >> 6, d = col & 63;
#pragma unroll
    for (int i = 0; i < 8; ++i) {
      const int M = rg * 16 + ((lane < 16) ? i : (i + 8));
      const int m = row0 + M;
      const int bb = m >> 12, n = m & (N_ - 1);
      qbf[((size_t)(bb * H_ + h) * N_ + n) * D_ + d] =
          f2bf((acc[j][i] + bias) * 0.125f);   // fold 1/sqrt(64)
    }
  }
}

// ---------------------------------------------------------------------------
// Kernel 6: attention. One block = (b,h) x 128 query rows; K/V^T DMA'd to LDS
// via TDM once per block. Per wave: S = Q*K^T (5x2 WMMA), 16-lane shfl
// softmax, P (bf16, K padded to 96) staged in LDS, O = P*V (3x4 WMMA).
// ---------------------------------------------------------------------------
__global__ void __launch_bounds__(256)
attention_kernel(const u16* __restrict__ qbf, const u16* __restrict__ kfull,
                 const u16* __restrict__ vT, u16* __restrict__ aout) {
  __shared__ __align__(16) u16 Kf[KK_ * D_];        // 80 x 64 row-major
  __shared__ __align__(16) u16 Vt[D_ * KKP_];       // 64 x 96 (v_full^T, padded)
  __shared__ __align__(16) u16 P[8 * 16 * KKP_];    // per-wave 16 x 96 P tiles
  const int bh = blockIdx.x >> 5;
  const int nt = blockIdx.x & 31;
#if HAVE_TDM
  if (threadIdx.x < 32) {
    tdm_load_lds(Kf, kfull + (size_t)bh * KK_ * D_, KK_ * D_ * 2);
    tdm_load_lds(Vt, vT + (size_t)bh * D_ * KKP_, D_ * KKP_ * 2);
    tdm_wait();
  }
#else
  for (int i = threadIdx.x; i < KK_ * D_; i += 256) Kf[i] = kfull[(size_t)bh * KK_ * D_ + i];
  for (int i = threadIdx.x; i < D_ * KKP_; i += 256) Vt[i] = vT[(size_t)bh * D_ * KKP_ + i];
#endif
  __syncthreads();

  const int lane = threadIdx.x & 31, w = threadIdx.x >> 5;
  const int mr = lane & 15, kh = lane >> 4;
  const int b = bh / H_, h = bh % H_;
  const int n0 = nt * 128 + w * 16;

  const u16* qrow = qbf + ((size_t)bh * N_ + n0 + mr) * D_;
  v16bf qa0 = load_frag_a(qrow, kh * 8);
  v16bf qa1 = load_frag_a(qrow, 32 + kh * 8);

  v8f s[5];
#pragma unroll
  for (int j = 0; j < 5; ++j) s[j] = v8f_zero();
#pragma unroll
  for (int j = 0; j < 5; ++j) {
    const u16* kr = Kf + (size_t)(j * 16 + mr) * D_;
    s[j] = wmma_bf16(qa0, load_frag_b(kr, kh * 16), s[j]);
    s[j] = wmma_bf16(qa1, load_frag_b(kr, 32 + kh * 16), s[j]);
  }

  // softmax over 80 ctx entries; row M lives across 16 lanes of one VGPR
  u16* pbase = P + (size_t)w * 16 * KKP_;
#pragma unroll
  for (int i = 0; i < 8; ++i) {
    float m = s[0][i];
#pragma unroll
    for (int j = 1; j < 5; ++j) m = fmaxf(m, s[j][i]);
    m = fmaxf(m, __shfl_xor(m, 1));
    m = fmaxf(m, __shfl_xor(m, 2));
    m = fmaxf(m, __shfl_xor(m, 4));
    m = fmaxf(m, __shfl_xor(m, 8));
    float pj[5], sum = 0.f;
#pragma unroll
    for (int j = 0; j < 5; ++j) { pj[j] = __expf(s[j][i] - m); sum += pj[j]; }
    sum += __shfl_xor(sum, 1);
    sum += __shfl_xor(sum, 2);
    sum += __shfl_xor(sum, 4);
    sum += __shfl_xor(sum, 8);
    const float inv = 1.0f / sum;
    const int M = (lane < 16) ? i : (i + 8);
#pragma unroll
    for (int j = 0; j < 5; ++j)
      pbase[M * KKP_ + j * 16 + mr] = f2bf(pj[j] * inv);
    pbase[M * KKP_ + KK_ + mr] = 0;           // zero pad cols 80..95
  }
  __syncthreads();

  v8f o[4];
#pragma unroll
  for (int j = 0; j < 4; ++j) o[j] = v8f_zero();
  const u16* prow = pbase + mr * KKP_;
#pragma unroll
  for (int ks = 0; ks < KKP_ / 32; ++ks) {
    v16bf pa = load_frag_a(prow, ks * 32 + kh * 8);
#pragma unroll
    for (int j = 0; j < 4; ++j)
      o[j] = wmma_bf16(pa, load_frag_b(Vt + (size_t)(j * 16 + mr) * KKP_,
                                       ks * 32 + kh * 16), o[j]);
  }
#pragma unroll
  for (int j = 0; j < 4; ++j) {
    const int col = h * D_ + j * 16 + mr;
#pragma unroll
    for (int i = 0; i < 8; ++i) {
      const int M = (lane < 16) ? i : (i + 8);
      aout[((size_t)b * N_ + n0 + M) * C_ + col] = f2bf(o[j][i]);
    }
  }
}

// ---------------------------------------------------------------------------
// Kernel 7: out-proj (65536 x 768 x 768), 32-row blocks, fused LayerNorm.
// Row stats combined across the 4 col-group waves with ds_add_f32 atomics;
// normalized output written straight from the accumulators (no fp32 tile).
// ---------------------------------------------------------------------------
__global__ void __launch_bounds__(256)
outproj_ln_kernel(const u16* __restrict__ abf, const u16* __restrict__ WpT,
                  const float* __restrict__ b_proj, const float* __restrict__ gamma,
                  const float* __restrict__ beta, float* __restrict__ out) {
  __shared__ __align__(16) u16 Atile[32 * C_];
  __shared__ float rsum[32];
  __shared__ float rsq[32];
  const int row0 = blockIdx.x * 32;
  const u16* src = abf + (size_t)row0 * C_;
#if HAVE_TDM
  if (threadIdx.x < 32) { tdm_load_lds(Atile, src, 32 * C_ * 2); tdm_wait(); }
#else
  for (int i = threadIdx.x; i < 32 * C_; i += 256) Atile[i] = src[i];
#endif
  if (threadIdx.x < 32) { rsum[threadIdx.x] = 0.f; rsq[threadIdx.x] = 0.f; }
  __syncthreads();

  const int lane = threadIdx.x & 31, w = threadIdx.x >> 5;
  const int mr = lane & 15, kh = lane >> 4;
  const int rg = w >> 2;                      // row group (0..1)
  const int cg = w & 3;                       // col group (0..3)
  const u16* arow = Atile + (size_t)(rg * 16 + mr) * C_;

  v8f acc[12];
#pragma unroll
  for (int j = 0; j < 12; ++j) acc[j] = v8f_zero();

  for (int ks = 0; ks < C_ / 32; ++ks) {
    v16bf a = load_frag_a(arow, ks * 32 + kh * 8);
#pragma unroll
    for (int j = 0; j < 12; ++j) {
      const int n = (cg * 12 + j) * 16 + mr;
      v16bf bm = load_frag_b(WpT + (size_t)n * C_, ks * 32 + kh * 16);
      acc[j] = wmma_bf16(a, bm, acc[j]);
    }
  }
  // add bias, accumulate per-row partial sums over this wave's 192 columns
#pragma unroll
  for (int i = 0; i < 8; ++i) {
    float s = 0.f, q = 0.f;
#pragma unroll
    for (int j = 0; j < 12; ++j) {
      const int col = (cg * 12 + j) * 16 + mr;
      const float v = acc[j][i] + b_proj[col];
      acc[j][i] = v;
      s += v; q += v * v;
    }
    s += __shfl_xor(s, 1); q += __shfl_xor(q, 1);
    s += __shfl_xor(s, 2); q += __shfl_xor(q, 2);
    s += __shfl_xor(s, 4); q += __shfl_xor(q, 4);
    s += __shfl_xor(s, 8); q += __shfl_xor(q, 8);
    const int M = rg * 16 + ((lane < 16) ? i : (i + 8));
    if (mr == 0) {                 // one lane per 16-group commits the row stat
      atomicAdd(&rsum[M], s);      // ds_add_f32
      atomicAdd(&rsq[M], q);
    }
  }
  __syncthreads();

  // normalize and write fp32 output directly from accumulators
#pragma unroll
  for (int i = 0; i < 8; ++i) {
    const int M = rg * 16 + ((lane < 16) ? i : (i + 8));
    const float mu  = rsum[M] * (1.0f / C_);
    const float var = rsq[M] * (1.0f / C_) - mu * mu;
    const float rs  = rsqrtf(var + 1e-5f);
#pragma unroll
    for (int j = 0; j < 12; ++j) {
      const int col = (cg * 12 + j) * 16 + mr;
      out[(size_t)(row0 + M) * C_ + col] =
          (acc[j][i] - mu) * rs * gamma[col] + beta[col];
    }
  }
}

// ---------------------------------------------------------------------------
// Host launch
// ---------------------------------------------------------------------------
extern "C" void kernel_launch(void* const* d_in, const int* in_sizes, int n_in,
                              void* d_out, int out_size, void* d_ws, size_t ws_size,
                              hipStream_t stream) {
  const float* x      = (const float*)d_in[0];
  const float* W_qkv  = (const float*)d_in[1];
  const float* b_qkv  = (const float*)d_in[2];
  const float* E_k    = (const float*)d_in[3];
  const float* E_v    = (const float*)d_in[4];
  const float* gk     = (const float*)d_in[5];
  const float* gv     = (const float*)d_in[6];
  const float* W_proj = (const float*)d_in[7];
  const float* b_proj = (const float*)d_in[8];
  const float* gamma  = (const float*)d_in[9];
  const float* beta   = (const float*)d_in[10];
  float* out = (float*)d_out;

  char* ws = (char*)d_ws;
  size_t off = 0;
  auto take = [&](size_t bytes) {
    char* p = ws + off;
    off += (bytes + 255) & ~(size_t)255;
    return p;
  };
  u16* xbf   = (u16*)take((size_t)B_ * N_ * C_ * 2);
  u16* xpbf  = (u16*)take((size_t)B_ * LSEQ_ * C_ * 2);
  u16* WqT   = (u16*)take((size_t)C_ * C_ * 2);
  u16* WkvT  = (u16*)take((size_t)2 * C_ * C_ * 2);
  u16* WpT   = (u16*)take((size_t)C_ * C_ * 2);
  u16* EkT   = (u16*)take((size_t)LK_ * LSEQ_ * 2);
  u16* EvT   = (u16*)take((size_t)LK_ * LSEQ_ * 2);
  u16* kvbf  = (u16*)take((size_t)B_ * LSEQ_ * 2 * C_ * 2);
  u16* kfull = (u16*)take((size_t)B_ * H_ * KK_ * D_ * 2);
  u16* vT    = (u16*)take((size_t)B_ * H_ * D_ * KKP_ * 2);
  u16* qbf   = (u16*)take((size_t)B_ * H_ * N_ * D_ * 2);
  u16* aoutb = (u16*)take((size_t)B_ * N_ * C_ * 2);
  (void)in_sizes; (void)n_in; (void)out_size; (void)ws_size;

  // 1. weights -> transposed bf16
  {
    const size_t tot = (size_t)C_ * C_ + 2 * C_ * C_ + C_ * C_ + 2 * LK_ * LSEQ_;
    prep_weights_kernel<<<dim3((unsigned)((tot + 255) / 256)), dim3(256), 0, stream>>>(
        W_qkv, W_proj, E_k, E_v, WqT, WkvT, WpT, EkT, EvT);
  }
  // 2. x -> bf16 + pooled first-128 tokens
  {
    const size_t tot = (size_t)B_ * N_ * C_ + (size_t)B_ * LSEQ_ * C_;
    convert_pool_kernel<<<dim3((unsigned)((tot + 255) / 256)), dim3(256), 0, stream>>>(
        x, xbf, xpbf);
  }
  // 3. pooled KV projection
  kvproj_kernel<<<dim3(B_ * LSEQ_ / 16), dim3(256), 0, stream>>>(xpbf, WkvT, b_qkv, kvbf);
  // 4. E^T compression + bank append
  kv_compress_kernel<<<dim3(B_ * H_), dim3(256), 0, stream>>>(kvbf, EkT, EvT, gk, gv,
                                                              kfull, vT);
  // 5. Q projection (scale folded), 32-row blocks
  qproj_kernel<<<dim3(B_ * N_ / 32), dim3(256), 0, stream>>>(xbf, WqT, b_qkv, qbf);
  // 6. attention (ctx = 80)
  attention_kernel<<<dim3(B_ * H_ * (N_ / 128)), dim3(256), 0, stream>>>(qbf, kfull, vT,
                                                                         aoutb);
  // 7. out-proj + LayerNorm, 32-row blocks
  outproj_ln_kernel<<<dim3(B_ * N_ / 32), dim3(256), 0, stream>>>(aoutb, WpT, b_proj,
                                                                  gamma, beta, out);
}
